// TriangleMultiplicationOutgoing_23708219474362
// MI455X (gfx1250) — matrix operand order repaired and checked by reading, not compiled, not run on hardware
//
#include <hip/hip_runtime.h>
#include <hip/hip_bf16.h>

// TriangleMultiplicationOutgoing for MI455X (gfx1250, wave32, WMMA bf16).
// All matmuls via v_wmma_f32_16x16x32_bf16. left/right stored channel-major
// bf16 (134MB combined -> L2-resident at 192MB) so the triangle einsum streams
// its tiles with global_load_async_to_lds_b128 (double-buffered, ASYNCcnt).

constexpr int kN    = 512;
constexpr int kNPos = kN * kN;           // 262144

typedef __attribute__((ext_vector_type(16))) __bf16 v16bf;
typedef __attribute__((ext_vector_type(8)))  __bf16 v8bf;
typedef __attribute__((ext_vector_type(4)))  __bf16 v4bf;
typedef __attribute__((ext_vector_type(8)))  float  v8f;

union V16U { v16bf v; v8bf h[2]; };

__device__ __forceinline__ v8f wmma_bf16(v16bf a, v16bf b, v8f c) {
  // (neg_a, A, neg_b, B, c_mod, C, reuse_a, reuse_b)
  return __builtin_amdgcn_wmma_f32_16x16x32_bf16(false, a, false, b, (short)0, c,
                                                 false, false);
}

// A fragment (16x32 bf16, M x K). rowp = &tile[lane&15][0].
// Lane<16: K in {k0..k0+7, k0+16..k0+23}; lane>=16: +8. Two 16B contiguous loads.
__device__ __forceinline__ v16bf load_a(const __bf16* rowp, int k0, int lane) {
  const int base = k0 + ((lane & 16) ? 8 : 0);
  V16U u;
  u.h[0] = *(const v8bf*)(rowp + base);
  u.h[1] = *(const v8bf*)(rowp + base + 16);
  return u.v;
}

// B fragment (32x16 bf16, K x N). colp = &W[n][0], K contiguous.
// Lane<16: K in {k0..k0+7, k0+16..k0+23}; lane>=16: +8. Two 16B contiguous loads.
__device__ __forceinline__ v16bf load_b(const __bf16* colp, int k0, int lane) {
  const int koff = k0 + ((lane & 16) ? 8 : 0);
  V16U u;
  u.h[0] = *(const v8bf*)(colp + koff);
  u.h[1] = *(const v8bf*)(colp + koff + 16);
  return u.v;
}

// gfx1250 async copy: 16B global -> LDS, tracked by ASYNCcnt.
__device__ __forceinline__ void async16(void* lds, const void* g) {
  const unsigned loff = (unsigned)(uintptr_t)lds;   // flat LDS addr low 32 bits
  asm volatile("global_load_async_to_lds_b128 %0, %1, off"
               :: "v"(loff), "v"(g) : "memory");
}
__device__ __forceinline__ void wait_async0() {
  asm volatile("s_wait_asynccnt 0x0" ::: "memory");
}
__device__ __forceinline__ void wait_async2() {
  asm volatile("s_wait_asynccnt 0x2" ::: "memory");
}

__device__ __forceinline__ float sigmoidf(float x) {
  return 1.0f / (1.0f + __expf(-x));
}

__device__ __forceinline__ float wsum32(float v) {
#pragma unroll
  for (int m = 1; m < 32; m <<= 1) v += __shfl_xor(v, m, 32);
  return v;
}

// ---------------------------------------------------------------------------
// Kernel 0: weights -> bf16, K-contiguous per output column.
// WtAll[640][128]: rows 0-127=Wl, 128-255=Wr, 256-383=Wlg, 384-511=Wrg,
// 512-639=Wog.  WoT[d][c] = Wout[c][d].
// ---------------------------------------------------------------------------
__global__ __launch_bounds__(256) void k_prep(
    const float* __restrict__ Wl, const float* __restrict__ Wr,
    const float* __restrict__ Wlg, const float* __restrict__ Wrg,
    const float* __restrict__ Wog, const float* __restrict__ Wout,
    __bf16* __restrict__ WtAll, __bf16* __restrict__ WoT) {
  int idx = blockIdx.x * 256 + threadIdx.x;
  if (idx < 640 * 128) {
    const int n = idx >> 7, k = idx & 127;
    const int r = n >> 7, nn = n & 127;
    const float* W = (r == 0) ? Wl : (r == 1) ? Wr : (r == 2) ? Wlg
                   : (r == 3) ? Wrg : Wog;
    WtAll[idx] = (__bf16)W[k * 128 + nn];
  } else {
    idx -= 640 * 128;
    if (idx < 128 * 128) {
      const int d = idx >> 7, c = idx & 127;
      WoT[idx] = (__bf16)Wout[c * 128 + d];
    }
  }
}

// ---------------------------------------------------------------------------
// Kernel 1: LayerNorm1 + 5 fused projections + sigmoid gating.
// 256 threads = 8 waves, 128 positions/block.
// left/right written CHANNEL-MAJOR (leftT[c][pos]) via an LDS transpose so the
// triangle kernel can async-DMA contiguous rows. gate stays position-major.
// ---------------------------------------------------------------------------
__global__ __launch_bounds__(256) void k_ln_proj(
    const float* __restrict__ Zr, const float* __restrict__ g1,
    const float* __restrict__ b1, const __bf16* __restrict__ Wt,
    const float* __restrict__ bl, const float* __restrict__ br,
    const float* __restrict__ blg, const float* __restrict__ brg,
    const float* __restrict__ bog, __bf16* __restrict__ leftT,
    __bf16* __restrict__ rightT, __bf16* __restrict__ gate) {
  __shared__ __bf16 Xs[128][136];  // normalized activations (A operand)
  __shared__ __bf16 Ys[128][136];  // output staging for the transpose
  const int lane = threadIdx.x & 31;
  const int wave = threadIdx.x >> 5;
  const int posBase = blockIdx.x * 128;

  // ---- phase A: layernorm (1 wave per row, 4 elems per lane) ----
  const float4 gv = *(const float4*)(g1 + lane * 4);
  const float4 bv = *(const float4*)(b1 + lane * 4);
  const float ga[4] = {gv.x, gv.y, gv.z, gv.w};
  const float ba[4] = {bv.x, bv.y, bv.z, bv.w};
  for (int r = wave; r < 128; r += 8) {
    const size_t pos = (size_t)posBase + r;
    const float4 x4 = *(const float4*)(Zr + pos * 128 + lane * 4);
    const float xa[4] = {x4.x, x4.y, x4.z, x4.w};
    float s = xa[0] + xa[1] + xa[2] + xa[3];
    float s2 = xa[0] * xa[0] + xa[1] * xa[1] + xa[2] * xa[2] + xa[3] * xa[3];
    s = wsum32(s);
    s2 = wsum32(s2);
    const float mu = s * (1.0f / 128.0f);
    const float var = s2 * (1.0f / 128.0f) - mu * mu;
    const float rs = rsqrtf(var + 1e-5f);
    v4bf y4;
#pragma unroll
    for (int e = 0; e < 4; ++e) y4[e] = (__bf16)((xa[e] - mu) * rs * ga[e] + ba[e]);
    *(v4bf*)&Xs[r][lane * 4] = y4;
  }
  __syncthreads();

  const int m0 = wave * 16;
  const __bf16* Arow = &Xs[m0 + (lane & 15)][0];
  const int nn = lane & 15;

  // ---- left = (Z@Wl+bl) * sigmoid(Z@Wlg+blg), staged then transposed ----
#pragma unroll 1
  for (int t = 0; t < 8; ++t) {
    v8f accM = {}, accG = {};
    const __bf16* BM = Wt + (size_t)(0 * 128 + t * 16 + nn) * 128;
    const __bf16* BG = Wt + (size_t)(2 * 128 + t * 16 + nn) * 128;
#pragma unroll
    for (int k0 = 0; k0 < 128; k0 += 32) {
      const v16bf a = load_a(Arow, k0, lane);
      accM = wmma_bf16(a, load_b(BM, k0, lane), accM);
      accG = wmma_bf16(a, load_b(BG, k0, lane), accG);
    }
    const int c = t * 16 + nn;
    const float bm = bl[c], bg = blg[c];
#pragma unroll
    for (int v = 0; v < 8; ++v) {
      const int m = v + ((lane & 16) ? 8 : 0);
      Ys[m0 + m][c] = (__bf16)((accM[v] + bm) * sigmoidf(accG[v] + bg));
    }
  }
  __syncthreads();
  for (int ch = wave; ch < 128; ch += 8) {   // coalesced channel-major write
    v4bf y;
#pragma unroll
    for (int e = 0; e < 4; ++e) y[e] = Ys[lane * 4 + e][ch];
    *(v4bf*)(leftT + (size_t)ch * kNPos + posBase + lane * 4) = y;
  }
  __syncthreads();

  // ---- right = (Z@Wr+br) * sigmoid(Z@Wrg+brg), staged then transposed ----
#pragma unroll 1
  for (int t = 0; t < 8; ++t) {
    v8f accM = {}, accG = {};
    const __bf16* BM = Wt + (size_t)(1 * 128 + t * 16 + nn) * 128;
    const __bf16* BG = Wt + (size_t)(3 * 128 + t * 16 + nn) * 128;
#pragma unroll
    for (int k0 = 0; k0 < 128; k0 += 32) {
      const v16bf a = load_a(Arow, k0, lane);
      accM = wmma_bf16(a, load_b(BM, k0, lane), accM);
      accG = wmma_bf16(a, load_b(BG, k0, lane), accG);
    }
    const int c = t * 16 + nn;
    const float bm = br[c], bg = brg[c];
#pragma unroll
    for (int v = 0; v < 8; ++v) {
      const int m = v + ((lane & 16) ? 8 : 0);
      Ys[m0 + m][c] = (__bf16)((accM[v] + bm) * sigmoidf(accG[v] + bg));
    }
  }
  __syncthreads();
  for (int ch = wave; ch < 128; ch += 8) {
    v4bf y;
#pragma unroll
    for (int e = 0; e < 4; ++e) y[e] = Ys[lane * 4 + e][ch];
    *(v4bf*)(rightT + (size_t)ch * kNPos + posBase + lane * 4) = y;
  }

  // ---- gate = sigmoid(Z@Wog + bog), position-major direct store ----
#pragma unroll 1
  for (int t = 0; t < 8; ++t) {
    v8f acc = {};
    const __bf16* BO = Wt + (size_t)(4 * 128 + t * 16 + nn) * 128;
#pragma unroll
    for (int k0 = 0; k0 < 128; k0 += 32)
      acc = wmma_bf16(load_a(Arow, k0, lane), load_b(BO, k0, lane), acc);
    const int c = t * 16 + nn;
    const float bo = bog[c];
#pragma unroll
    for (int v = 0; v < 8; ++v) {
      const int m = v + ((lane & 16) ? 8 : 0);
      const size_t pos = (size_t)posBase + m0 + m;
      gate[pos * 128 + c] = (__bf16)sigmoidf(acc[v] + bo);
    }
  }
}

// ---------------------------------------------------------------------------
// Kernel 2: triangle einsum ab[i,j,c] = sum_k L[i,k,c]*R[j,k,c].
// grid = (64 tiles of 64x64, 128 channels). Tiles streamed from the
// channel-major bf16 operands with global_load_async_to_lds_b128,
// double-buffered on ASYNCcnt. Stores abT[c][i*512+j] (f32, coalesced).
// ---------------------------------------------------------------------------
__global__ __launch_bounds__(256) void k_tri(const __bf16* __restrict__ leftT,
                                             const __bf16* __restrict__ rightT,
                                             float* __restrict__ abT) {
  __shared__ __bf16 Lt[2][64][40];
  __shared__ __bf16 Rt[2][64][40];
  const int lane = threadIdx.x & 31;
  const int wave = threadIdx.x >> 5;
  const int c = blockIdx.y;
  const int i0 = (blockIdx.x >> 3) * 64;
  const int j0 = (blockIdx.x & 7) * 64;

  // each thread DMAs one 16B segment of one tile row per buffer
  const int r  = threadIdx.x >> 2;         // 0..63
  const int sg = (threadIdx.x & 3) * 8;    // bf16 offset {0,8,16,24}
  const __bf16* Lsrc = leftT  + (size_t)c * kNPos + (size_t)(i0 + r) * 512 + sg;
  const __bf16* Rsrc = rightT + (size_t)c * kNPos + (size_t)(j0 + r) * 512 + sg;

  const int ti  = wave >> 1;         // A row-strip (shared by this wave's 2 tiles)
  const int tj0 = (wave & 1) * 2;    // two adjacent 16-col B tiles
  v8f acc0 = {}, acc1 = {};

  async16(&Lt[0][r][sg], Lsrc);      // prologue: k0 = 0 into buffer 0
  async16(&Rt[0][r][sg], Rsrc);

  int buf = 0;
  for (int k0 = 0; k0 < 512; k0 += 32) {
    __syncthreads();                 // everyone done reading buf^1
    if (k0 + 32 < 512) {
      async16(&Lt[buf ^ 1][r][sg], Lsrc + k0 + 32);
      async16(&Rt[buf ^ 1][r][sg], Rsrc + k0 + 32);
      wait_async2();                 // in-order: first 2 (buf) are complete
    } else {
      wait_async0();
    }
    __syncthreads();                 // buf's data visible to all waves
    const v16bf a = load_a(&Lt[buf][ti * 16 + (lane & 15)][0], 0, lane);
    acc0 = wmma_bf16(a, load_b(&Rt[buf][tj0 * 16 + (lane & 15)][0], 0, lane), acc0);
    acc1 = wmma_bf16(a, load_b(&Rt[buf][(tj0 + 1) * 16 + (lane & 15)][0], 0, lane), acc1);
    buf ^= 1;
  }
#pragma unroll
  for (int v = 0; v < 8; ++v) {
    const int m = v + ((lane & 16) ? 8 : 0);
    const int nn = lane & 15;
    const size_t row = (size_t)(i0 + ti * 16 + m) * 512;
    abT[(size_t)c * kNPos + row + j0 + tj0 * 16 + nn] = acc0[v];
    abT[(size_t)c * kNPos + row + j0 + (tj0 + 1) * 16 + nn] = acc1[v];
  }
}

// ---------------------------------------------------------------------------
// Kernel 3: LayerNorm2 + @Wout + sigmoid-gate + residual. 128 positions/block.
// ---------------------------------------------------------------------------
__global__ __launch_bounds__(256) void k_out(
    const float* __restrict__ abT, const float* __restrict__ g2,
    const float* __restrict__ b2, const __bf16* __restrict__ WoT,
    const float* __restrict__ obias, const __bf16* __restrict__ gate,
    const float* __restrict__ Zr, float* __restrict__ out) {
  __shared__ float Sab[128][132];
  __shared__ __bf16 Xb[128][136];
  const int lane = threadIdx.x & 31;
  const int wave = threadIdx.x >> 5;
  const int pBase = blockIdx.x * 128;

  {  // gather all 128 channels for 128 positions (coalesced per channel)
    const int row = threadIdx.x & 127;
    const int cb = (threadIdx.x >> 7) * 64;
    for (int cc = 0; cc < 64; ++cc) {
      const int ch = cb + cc;
      Sab[row][ch] = abT[(size_t)ch * kNPos + pBase + row];
    }
  }
  __syncthreads();

  const float4 gv = *(const float4*)(g2 + lane * 4);
  const float4 bv = *(const float4*)(b2 + lane * 4);
  const float ga[4] = {gv.x, gv.y, gv.z, gv.w};
  const float ba[4] = {bv.x, bv.y, bv.z, bv.w};
  for (int r = wave; r < 128; r += 8) {
    const float* rp = &Sab[r][lane * 4];
    const float xa[4] = {rp[0], rp[1], rp[2], rp[3]};
    float s = xa[0] + xa[1] + xa[2] + xa[3];
    float s2 = xa[0] * xa[0] + xa[1] * xa[1] + xa[2] * xa[2] + xa[3] * xa[3];
    s = wsum32(s);
    s2 = wsum32(s2);
    const float mu = s * (1.0f / 128.0f);
    const float var = s2 * (1.0f / 128.0f) - mu * mu;
    const float rs = rsqrtf(var + 1e-5f);
    v4bf y4;
#pragma unroll
    for (int e = 0; e < 4; ++e) y4[e] = (__bf16)((xa[e] - mu) * rs * ga[e] + ba[e]);
    *(v4bf*)&Xb[r][lane * 4] = y4;
  }
  __syncthreads();

  const int m0 = wave * 16;
  const __bf16* Arow = &Xb[m0 + (lane & 15)][0];
  const int nn = lane & 15;
#pragma unroll 1
  for (int t = 0; t < 8; ++t) {
    v8f acc = {};
    const __bf16* Bp = WoT + (size_t)(t * 16 + nn) * 128;
#pragma unroll
    for (int k0 = 0; k0 < 128; k0 += 32)
      acc = wmma_bf16(load_a(Arow, k0, lane), load_b(Bp, k0, lane), acc);
    const int d = t * 16 + nn;
    const float ob = obias[d];
#pragma unroll
    for (int v = 0; v < 8; ++v) {
      const int m = v + ((lane & 16) ? 8 : 0);
      const size_t p = (size_t)pBase + m0 + m;
      const float gt = (float)gate[p * 128 + d];
      out[p * 128 + d] = Zr[p * 128 + d] + gt * (acc[v] + ob);
    }
  }
}

// ---------------------------------------------------------------------------
extern "C" void kernel_launch(void* const* d_in, const int* in_sizes, int n_in,
                              void* d_out, int out_size, void* d_ws,
                              size_t ws_size, hipStream_t stream) {
  const float* Zr    = (const float*)d_in[0];
  const float* ln1_g = (const float*)d_in[1];
  const float* ln1_b = (const float*)d_in[2];
  const float* Wl    = (const float*)d_in[3];
  const float* bl    = (const float*)d_in[4];
  const float* Wr    = (const float*)d_in[5];
  const float* br    = (const float*)d_in[6];
  const float* Wlg   = (const float*)d_in[7];
  const float* blg   = (const float*)d_in[8];
  const float* Wrg   = (const float*)d_in[9];
  const float* brg   = (const float*)d_in[10];
  const float* Wog   = (const float*)d_in[11];
  const float* bog   = (const float*)d_in[12];
  const float* ln2_g = (const float*)d_in[13];
  const float* ln2_b = (const float*)d_in[14];
  const float* Wout  = (const float*)d_in[15];
  const float* obias = (const float*)d_in[16];
  float* out = (float*)d_out;

  // workspace layout (bytes), all 256B aligned:
  char* ws = (char*)d_ws;
  __bf16* WtAll  = (__bf16*)(ws);                       // 640*128*2 = 160 KB
  __bf16* WoT    = (__bf16*)(ws + 163840);              // 128*128*2 =  32 KB
  __bf16* leftT  = (__bf16*)(ws + 196608);              // 64 MB, [c][pos]
  __bf16* rightT = (__bf16*)(ws + 196608 + 67108864ULL);
  __bf16* gate   = (__bf16*)(ws + 196608 + 2ULL * 67108864ULL);  // [pos][c]
  float*  abT    = (float*)(ws + 196608 + 3ULL * 67108864ULL);   // 128 MB [c][pos]

  k_prep<<<384, 256, 0, stream>>>(Wl, Wr, Wlg, Wrg, Wog, Wout, WtAll, WoT);
  k_ln_proj<<<kNPos / 128, 256, 0, stream>>>(Zr, ln1_g, ln1_b, WtAll, bl, br,
                                             blg, brg, bog, leftT, rightT, gate);
  k_tri<<<dim3(64, 128), 256, 0, stream>>>(leftT, rightT, abT);
  k_out<<<kNPos / 128, 256, 0, stream>>>(abT, ln2_g, ln2_b, WoT, obias, gate,
                                         Zr, out);
}
